// Gpsa2RoIPool_90331752169715
// MI455X (gfx1250) — matrix operand order repaired and checked by reading, not compiled, not run on hardware
//
#include <hip/hip_runtime.h>
#include <hip/hip_bf16.h>

typedef _Float16 f16;
typedef __attribute__((ext_vector_type(16))) _Float16 v16h;
typedef __attribute__((ext_vector_type(8)))  _Float16 v8h;
typedef __attribute__((ext_vector_type(8)))  float    v8f;

#define WMMA(a,b,c) __builtin_amdgcn_wmma_f32_16x16x32_f16(false,(a),false,(b),(short)0,(c),false,false)

union V16U { v16h v; v8h h[2]; };

// A fragment (16x32 f16, MxK): lane l -> row l%16; halves e -> K = e + 8*((e>=8)+lhi)
// => two contiguous 8-half (16B) runs at cols 8*lhi and 16+8*lhi of the row.
__device__ __forceinline__ v16h frag_a(const f16* rowp, int lhi) {
  V16U u;
  u.h[0] = *(const v8h*)(rowp + 8 * lhi);
  u.h[1] = *(const v8h*)(rowp + 16 + 8 * lhi);
  return u.v;
}
// B fragment (32x16 f16, KxN): lane l -> col l%16; halves e -> K = e + 8*((e>=8)+lhi)
// => two contiguous 8-half (16B) runs at K 8*lhi and 16+8*lhi for this lane's column.
__device__ __forceinline__ v16h frag_b(const f16* rowp, int lhi) {
  V16U u;
  u.h[0] = *(const v8h*)(rowp + 8 * lhi);
  u.h[1] = *(const v8h*)(rowp + 16 + 8 * lhi);
  return u.v;
}

// ---------------- geometry constants ----------------
#define NROIS 512
#define EDIM  256
#define FHH   152
#define FWW   256
#define NQ    49
#define SS    196
#define SPAD  208   // 196 padded to 13*16
#define VPAD  224   // 196 padded to 7*32 for the PV GEMM K dim
#define HEADS 8
#define DH    32

// ---------------- prep kernels ----------------
// features [1,256,152,256] f32 -> channel-last f16 [y][x][c] for coalesced gathers
__global__ void prep_feat_kernel(const float* __restrict__ f, f16* __restrict__ o) {
  int i = blockIdx.x * 256 + threadIdx.x;
  const int total = EDIM * FHH * FWW;
  if (i >= total) return;
  int c = i / (FHH * FWW);
  int yx = i % (FHH * FWW);
  o[yx * EDIM + c] = (f16)f[i];
}

__global__ void prep_w_kernel(const float* __restrict__ inw, const float* __restrict__ outw,
                              f16* __restrict__ wk, f16* __restrict__ wv, f16* __restrict__ wo) {
  int i = blockIdx.x * 256 + threadIdx.x;
  if (i < EDIM * EDIM) {
    wk[i] = (f16)inw[EDIM * EDIM + i];       // Wk rows 256..511
    wv[i] = (f16)inw[2 * EDIM * EDIM + i];   // Wv rows 512..767
    wo[i] = (f16)outw[i];
  }
}

// q = (cq @ Wq^T + bq) * 1/sqrt(DH), laid out [h][64(m pad)][32(d)] f16, pad rows zero
__global__ void prep_q_kernel(const float* __restrict__ cq, const float* __restrict__ inw,
                              const float* __restrict__ inb, f16* __restrict__ q16) {
  int m = blockIdx.x;       // 0..63
  int col = threadIdx.x;    // 0..255
  float v = 0.f;
  if (m < NQ) {
    float acc = 0.f;
    for (int e = 0; e < EDIM; e++) acc += cq[m * EDIM + e] * inw[col * EDIM + e];
    v = (acc + inb[col]) * 0.17677669529663687f;  // 1/sqrt(32)
  }
  int h = col >> 5, d = col & 31;
  q16[(h * 64 + m) * DH + d] = (f16)v;
}

// zero K rows s=196..207 and V cols s=196..223 (per roi,head)
__global__ void zero_pads_kernel(f16* __restrict__ Kws, f16* __restrict__ Vws) {
  int i = blockIdx.x * 256 + threadIdx.x;
  if (i < NROIS * HEADS * 12 * DH) {
    int rh = i / (12 * DH);
    int r = i % (12 * DH);
    int s = SS + r / DH, d = r & 31;
    Kws[(rh * SPAD + s) * DH + d] = (f16)0.f;
  }
  if (i < NROIS * HEADS * DH * 28) {
    int rh = i / (DH * 28);
    int r = i % (DH * 28);
    int d = r / 28, s = SS + r % 28;
    Vws[(rh * DH + d) * VPAD + s] = (f16)0.f;
  }
}

// ---------------- RoIAlign ----------------
__device__ __forceinline__ float bilin(const f16* __restrict__ feat, float y, float x, int c) {
  float valid = (y >= -1.0f && y <= (float)FHH && x >= -1.0f && x <= (float)FWW) ? 1.f : 0.f;
  y = fminf(fmaxf(y, 0.f), (float)(FHH - 1));
  x = fminf(fmaxf(x, 0.f), (float)(FWW - 1));
  float y0 = floorf(y), x0 = floorf(x);
  int y0i = (int)y0, x0i = (int)x0;
  int y1i = y0i + 1 < FHH ? y0i + 1 : FHH - 1;
  int x1i = x0i + 1 < FWW ? x0i + 1 : FWW - 1;
  float ly = y - y0, lx = x - x0, hy = 1.f - ly, hx = 1.f - lx;
  float f00 = (float)feat[(y0i * FWW + x0i) * EDIM + c];
  float f01 = (float)feat[(y0i * FWW + x1i) * EDIM + c];
  float f10 = (float)feat[(y1i * FWW + x0i) * EDIM + c];
  float f11 = (float)feat[(y1i * FWW + x1i) * EDIM + c];
  return valid * (hy * hx * f00 + hy * lx * f01 + ly * hx * f10 + ly * lx * f11);
}

// pos path: 7x7, scale 1/16, sr=2 -> d_out * balance. lane = channel (coalesced taps)
__global__ void __launch_bounds__(256) pos_kernel(const f16* __restrict__ feat,
                                                  const float* __restrict__ rois,
                                                  const float* __restrict__ balp,
                                                  float* __restrict__ out) {
  int n = blockIdx.x, c = threadIdx.x;
  float balance = fminf(fmaxf(balp[0] / 6.f + 0.5f, 0.f), 1.f);
  const float scale = 1.f / 16.f;
  float x1 = rois[n * 5 + 1], y1 = rois[n * 5 + 2], x2 = rois[n * 5 + 3], y2 = rois[n * 5 + 4];
  float sx = x1 * scale - 0.5f, sy = y1 * scale - 0.5f;
  float bw = (x2 - x1) * scale / 7.f, bh = (y2 - y1) * scale / 7.f;
  for (int q = 0; q < NQ; q++) {
    int oy = q / 7, ox = q % 7;
    float acc = 0.f;
    #pragma unroll
    for (int iy = 0; iy < 2; iy++)
      #pragma unroll
      for (int ix = 0; ix < 2; ix++) {
        float y = sy + ((float)oy + (iy + 0.5f) * 0.5f) * bh;
        float x = sx + ((float)ox + (ix + 0.5f) * 0.5f) * bw;
        acc += bilin(feat, y, x, c);
      }
    out[n * (EDIM * NQ) + c * NQ + q] = balance * 0.25f * acc;
  }
}

// cf path: 14x14, on roi*1/16 at scale 1.0, sr=1 -> cf16 [roi][208(s pad)][256]
__global__ void __launch_bounds__(256) cf_kernel(const f16* __restrict__ feat,
                                                 const float* __restrict__ rois,
                                                 f16* __restrict__ cf) {
  int n = blockIdx.x, c = threadIdx.x;
  const float scale = 1.f / 16.f;
  float x1 = rois[n * 5 + 1] * scale, y1 = rois[n * 5 + 2] * scale;
  float x2 = rois[n * 5 + 3] * scale, y2 = rois[n * 5 + 4] * scale;
  float sx = x1 - 0.5f, sy = y1 - 0.5f;
  float bw = (x2 - x1) / 14.f, bh = (y2 - y1) / 14.f;
  for (int s = 0; s < SS; s++) {
    int gy = s / 14, gx = s % 14;
    float y = sy + ((float)gy + 0.5f) * bh;
    float x = sx + ((float)gx + 0.5f) * bw;
    cf[(n * SPAD + s) * EDIM + c] = (f16)bilin(feat, y, x, c);
  }
}

// ---------------- K/V projection GEMM (WMMA) ----------------
// per roi: C[208(s) x 256(c)] = cf[208 x 256] @ W^T, for Wk and Wv.
// K stored [roi][h][s(208)][d(32)]; V stored transposed [roi][h][d(32)][s(224)]
__global__ void __launch_bounds__(256) proj_kernel(const f16* __restrict__ cf,
                                                   const f16* __restrict__ wk,
                                                   const f16* __restrict__ wv,
                                                   const float* __restrict__ inb,
                                                   f16* __restrict__ Kws, f16* __restrict__ Vws) {
  int roi = blockIdx.x;
  int lane = threadIdx.x & 31, wave = threadIdx.x >> 5;
  int lhi = lane >> 4, lmod = lane & 15;
  const f16* A = cf + roi * SPAD * EDIM;
  for (int t = wave; t < 13 * 16 * 2; t += 8) {
    int mat = t & 1;            // 0=K, 1=V
    int mt = (t >> 1) % 13;     // s tile
    int nt = (t >> 1) / 13;     // cout tile
    const f16* W = mat ? wv : wk;
    int arow = 16 * mt + lmod;
    int cout = 16 * nt + lmod;
    v8f c = {};
    #pragma unroll
    for (int kc = 0; kc < 8; kc++) {
      v16h a = frag_a(A + arow * EDIM + 32 * kc, lhi);
      v16h b = frag_b(W + cout * EDIM + 32 * kc, lhi);
      c = WMMA(a, b, c);
    }
    float bias = inb[(mat ? 2 * EDIM : EDIM) + cout];
    int h = cout >> 5, d = cout & 31;
    #pragma unroll
    for (int g = 0; g < 8; g++) {
      int s = 16 * mt + g + 8 * lhi;
      if (s < SS) {
        f16 v = (f16)(c[g] + bias);
        if (mat == 0) Kws[((roi * HEADS + h) * SPAD + s) * DH + d] = v;
        else          Vws[((roi * HEADS + h) * DH + d) * VPAD + s] = v;
      }
    }
  }
}

// ---------------- fused attention (WMMA + in-register softmax) ----------------
// block = one roi, wave = one head. Two-pass softmax with logits recompute;
// P tiles staged 64x32 f16 per wave through LDS (C-frag -> A-frag transpose).
__global__ void __launch_bounds__(256) attn_kernel(const f16* __restrict__ q16,
                                                   const f16* __restrict__ Kws,
                                                   const f16* __restrict__ Vws,
                                                   f16* __restrict__ Ows) {
  __shared__ f16 xp_all[HEADS][64 * 32];
  int roi = blockIdx.x;
  int lane = threadIdx.x & 31, h = threadIdx.x >> 5;
  int lhi = lane >> 4, lmod = lane & 15;
  f16* xp = xp_all[h];
  const f16* Q  = q16 + h * 64 * DH;
  const f16* Kb = Kws + (roi * HEADS + h) * SPAD * DH;
  const f16* Vb = Vws + (roi * HEADS + h) * DH * VPAD;

  v16h aq[4];
  #pragma unroll
  for (int mt = 0; mt < 4; mt++) aq[mt] = frag_a(Q + (16 * mt + lmod) * DH, lhi);

  float rmax[4][8];
  #pragma unroll
  for (int mt = 0; mt < 4; mt++)
    #pragma unroll
    for (int g = 0; g < 8; g++) rmax[mt][g] = -1e30f;

  // pass A: row maxima
  for (int nt = 0; nt < 13; nt++) {
    v16h bk = frag_b(Kb + (16 * nt + lmod) * DH, lhi);
    bool bad = (nt == 12) && (lmod >= 4);  // s >= 196
    #pragma unroll
    for (int mt = 0; mt < 4; mt++) {
      v8f c = {};
      c = WMMA(aq[mt], bk, c);
      #pragma unroll
      for (int g = 0; g < 8; g++) {
        float v = bad ? -1e30f : c[g];
        rmax[mt][g] = fmaxf(rmax[mt][g], v);
      }
    }
  }
  // reduce max across the 16-lane half holding a row's columns
  #pragma unroll
  for (int mt = 0; mt < 4; mt++)
    #pragma unroll
    for (int g = 0; g < 8; g++) {
      float m = rmax[mt][g];
      m = fmaxf(m, __shfl_xor(m, 1, 32));
      m = fmaxf(m, __shfl_xor(m, 2, 32));
      m = fmaxf(m, __shfl_xor(m, 4, 32));
      m = fmaxf(m, __shfl_xor(m, 8, 32));
      rmax[mt][g] = m;
    }

  float rsum[4][8];
  v8f oacc[4][2];
  #pragma unroll
  for (int mt = 0; mt < 4; mt++) {
    #pragma unroll
    for (int g = 0; g < 8; g++) rsum[mt][g] = 0.f;
    oacc[mt][0] = (v8f){};
    oacc[mt][1] = (v8f){};
  }

  // pass B: recompute logits, exp, stage P chunk (32 s-cols), PV accumulate
  for (int cch = 0; cch < 7; cch++) {
    #pragma unroll
    for (int sub = 0; sub < 2; sub++) {
      int nt = 2 * cch + sub;
      if (nt < 13) {
        v16h bk = frag_b(Kb + (16 * nt + lmod) * DH, lhi);
        bool bad = (nt == 12) && (lmod >= 4);
        #pragma unroll
        for (int mt = 0; mt < 4; mt++) {
          v8f c = {};
          c = WMMA(aq[mt], bk, c);
          #pragma unroll
          for (int g = 0; g < 8; g++) {
            float p = __expf(c[g] - rmax[mt][g]);
            if (bad) p = 0.f;
            rsum[mt][g] += p;
            xp[(16 * mt + g + 8 * lhi) * 32 + 16 * sub + lmod] = (f16)p;
          }
        }
      } else {  // s 208..223 pad -> zeros
        #pragma unroll
        for (int mt = 0; mt < 4; mt++)
          #pragma unroll
          for (int g = 0; g < 8; g++)
            xp[(16 * mt + g + 8 * lhi) * 32 + 16 * sub + lmod] = (f16)0.f;
      }
    }
    asm volatile("s_wait_dscnt 0" ::: "memory");  // LDS transpose visible within wave
    v16h bv0 = frag_b(Vb + (0  + lmod) * VPAD + 32 * cch, lhi);
    v16h bv1 = frag_b(Vb + (16 + lmod) * VPAD + 32 * cch, lhi);
    #pragma unroll
    for (int mt = 0; mt < 4; mt++) {
      v16h ap = frag_a(xp + (16 * mt + lmod) * 32, lhi);
      oacc[mt][0] = WMMA(ap, bv0, oacc[mt][0]);
      oacc[mt][1] = WMMA(ap, bv1, oacc[mt][1]);
    }
    asm volatile("s_wait_dscnt 0" ::: "memory");  // reads done before next chunk overwrites
  }

  // reduce sums, scale, store o (f16) with zeroed pad rows for out-proj
  #pragma unroll
  for (int mt = 0; mt < 4; mt++)
    #pragma unroll
    for (int g = 0; g < 8; g++) {
      float s = rsum[mt][g];
      s += __shfl_xor(s, 1, 32);
      s += __shfl_xor(s, 2, 32);
      s += __shfl_xor(s, 4, 32);
      s += __shfl_xor(s, 8, 32);
      float rinv = 1.f / s;
      int m = 16 * mt + g + 8 * lhi;
      #pragma unroll
      for (int dt = 0; dt < 2; dt++) {
        float val = oacc[mt][dt][g] * rinv;
        f16 ov = (m < NQ) ? (f16)val : (f16)0.f;
        Ows[(roi * 64 + m) * EDIM + h * DH + 16 * dt + lmod] = ov;
      }
    }
}

// ---------------- output projection GEMM + final combine ----------------
__global__ void __launch_bounds__(256) outproj_kernel(const f16* __restrict__ Ows,
                                                      const f16* __restrict__ wo,
                                                      const float* __restrict__ outb,
                                                      const float* __restrict__ balp,
                                                      float* __restrict__ out) {
  int roi = blockIdx.x;
  int lane = threadIdx.x & 31, wave = threadIdx.x >> 5;
  int lhi = lane >> 4, lmod = lane & 15;
  float balance = fminf(fmaxf(balp[0] / 6.f + 0.5f, 0.f), 1.f);
  float w1 = 1.f - balance;
  const f16* A = Ows + roi * 64 * EDIM;
  for (int t = wave; t < 64; t += 8) {
    int mt = t & 3, nt = t >> 2;
    v8f c = {};
    #pragma unroll
    for (int kc = 0; kc < 8; kc++) {
      v16h a = frag_a(A + (16 * mt + lmod) * EDIM + 32 * kc, lhi);
      v16h b = frag_b(wo + (16 * nt + lmod) * EDIM + 32 * kc, lhi);
      c = WMMA(a, b, c);
    }
    int cc = 16 * nt + lmod;
    float bias = outb[cc];
    #pragma unroll
    for (int g = 0; g < 8; g++) {
      int m = 16 * mt + g + 8 * lhi;
      if (m < NQ) {
        int idx = roi * (EDIM * NQ) + cc * NQ + m;
        out[idx] = out[idx] + (c[g] + bias) * w1;
      }
    }
  }
}

// ---------------- host launch ----------------
extern "C" void kernel_launch(void* const* d_in, const int* in_sizes, int n_in,
                              void* d_out, int out_size, void* d_ws, size_t ws_size,
                              hipStream_t stream) {
  const float* features = (const float*)d_in[0];
  const float* rois     = (const float*)d_in[1];
  const float* cq       = (const float*)d_in[2];
  const float* inw      = (const float*)d_in[3];
  const float* inb      = (const float*)d_in[4];
  const float* outw     = (const float*)d_in[5];
  const float* outb     = (const float*)d_in[6];
  const float* balp     = (const float*)d_in[7];
  float* out = (float*)d_out;

  // workspace carve (all 16B-aligned)
  char* ws = (char*)d_ws;
  size_t off = 0;
  f16* feat16 = (f16*)(ws + off); off += (size_t)FHH * FWW * EDIM * 2;          // 19.9 MB
  f16* wk16   = (f16*)(ws + off); off += (size_t)EDIM * EDIM * 2;
  f16* wv16   = (f16*)(ws + off); off += (size_t)EDIM * EDIM * 2;
  f16* wo16   = (f16*)(ws + off); off += (size_t)EDIM * EDIM * 2;
  f16* q16    = (f16*)(ws + off); off += (size_t)HEADS * 64 * DH * 2;
  f16* cf16   = (f16*)(ws + off); off += (size_t)NROIS * SPAD * EDIM * 2;       // 54.5 MB
  f16* K16    = (f16*)(ws + off); off += (size_t)NROIS * HEADS * SPAD * DH * 2; // 54.5 MB
  f16* V16    = (f16*)(ws + off); off += (size_t)NROIS * HEADS * DH * VPAD * 2; // 58.7 MB
  f16* o16    = (f16*)(ws + off); off += (size_t)NROIS * 64 * EDIM * 2;         // 16.8 MB

  const int featN = EDIM * FHH * FWW;
  prep_feat_kernel<<<(featN + 255) / 256, 256, 0, stream>>>(features, feat16);
  prep_w_kernel<<<(EDIM * EDIM + 255) / 256, 256, 0, stream>>>(inw, outw, wk16, wv16, wo16);
  prep_q_kernel<<<64, 256, 0, stream>>>(cq, inw, inb, q16);
  const int padN = NROIS * HEADS * DH * 28;  // max of the two pad regions
  zero_pads_kernel<<<(padN + 255) / 256, 256, 0, stream>>>(K16, V16);

  pos_kernel<<<NROIS, 256, 0, stream>>>(feat16, rois, balp, out);
  cf_kernel<<<NROIS, 256, 0, stream>>>(feat16, rois, cf16);
  proj_kernel<<<NROIS, 256, 0, stream>>>(cf16, wk16, wv16, inb, K16, V16);
  attn_kernel<<<NROIS, 256, 0, stream>>>(q16, K16, V16, o16);
  outproj_kernel<<<NROIS, 256, 0, stream>>>(o16, wo16, outb, balp, out);
}
